// GraphToVoxelNet_22497038697246
// MI455X (gfx1250) — matrix operand order repaired and checked by reading, not compiled, not run on hardware
//
#include <hip/hip_runtime.h>
#include <hip/hip_bf16.h>
#include <math.h>

// ---------------- CDNA5 WMMA types ----------------
typedef __attribute__((ext_vector_type(16))) __bf16 v16bf;
typedef __attribute__((ext_vector_type(8)))  float  v8f;

// =====================================================================
// f32-in / bf16-compute / f32-out GEMM.  One wave computes a 16(M)x64(N)
// strip: one A fragment, 4 independent WMMAs per 32-deep K step (fills the
// bf16-WMMA RAW hazard window and amortizes A traffic).  K is a compile-time
// constant (128 or 256) so the K-loop fully unrolls -> 16/32 static v_wmma.
// A and B fragments both use contiguous k order, k = 16*half + s (wave32).
// =====================================================================
template <int K, bool BIAS, bool RELU>
__global__ void k_gemm(const float* __restrict__ A, const float* __restrict__ B,
                       const float* __restrict__ bias, float* __restrict__ C,
                       int M, int N) {
    int wave   = blockIdx.x * (blockDim.x >> 5) + (threadIdx.x >> 5);
    int tilesN = N >> 6;                       // 64-wide wave tiles
    int totT   = (M >> 4) * tilesN;
    if (wave >= totT) return;                  // wave-uniform; EXEC stays all-ones
    int tm = (wave / tilesN) << 4;
    int tn = (wave % tilesN) << 6;

    int lane = threadIdx.x & 31;
    int half = lane >> 4;                      // 0: lanes 0-15, 1: lanes 16-31
    int r15  = lane & 15;                      // A row in tile / B,C column in subtile

    v8f acc[4] = {v8f{}, v8f{}, v8f{}, v8f{}};
    const float* Abase = A + (size_t)(tm + r15) * K;

#pragma unroll
    for (int kk = 0; kk < K; kk += 32) {
        const float* Arow = Abase + kk;
        if (kk + 32 < K) __builtin_prefetch(Arow + 32, 0, 0);  // global_prefetch_b8

        // A 16x32 bf16 fragment: per lane = two contiguous 8-float runs
        //   frag[0..7]  = Arow[half*16 + 0..7],  frag[8..15] = Arow[half*16 + 8 + 0..7]
        v16bf a;
        {
            const float4* p0 = (const float4*)(Arow + (half << 4));
            const float4* p1 = (const float4*)(Arow + 8 + (half << 4));
            float4 f0 = p0[0], f1 = p0[1], f2 = p1[0], f3 = p1[1];
            a[0]  = (__bf16)f0.x; a[1]  = (__bf16)f0.y; a[2]  = (__bf16)f0.z; a[3]  = (__bf16)f0.w;
            a[4]  = (__bf16)f1.x; a[5]  = (__bf16)f1.y; a[6]  = (__bf16)f1.z; a[7]  = (__bf16)f1.w;
            a[8]  = (__bf16)f2.x; a[9]  = (__bf16)f2.y; a[10] = (__bf16)f2.z; a[11] = (__bf16)f2.w;
            a[12] = (__bf16)f3.x; a[13] = (__bf16)f3.y; a[14] = (__bf16)f3.z; a[15] = (__bf16)f3.w;
        }

        // B 32x16 bf16 fragments: lanes 0-15 hold K=kk..kk+15, lanes 16-31 K=kk+16..kk+31
        const float* Brow = B + (size_t)(kk + (half << 4)) * N + tn + r15;
#pragma unroll
        for (int j = 0; j < 4; ++j) {
            const float* Bp = Brow + (j << 4);
            v16bf bf;
#pragma unroll
            for (int v = 0; v < 16; ++v)
                bf[v] = (__bf16)Bp[(size_t)v * N];
            acc[j] = __builtin_amdgcn_wmma_f32_16x16x32_bf16(false, a, false, bf,
                                                             (short)0, acc[j], false, false);
        }
    }

    // C 16x16 f32 per subtile: VGPR r -> M = r + half*8, N = r15
#pragma unroll
    for (int j = 0; j < 4; ++j) {
        int n = tn + (j << 4) + r15;
        float bv = BIAS ? bias[n] : 0.0f;
#pragma unroll
        for (int r = 0; r < 8; ++r) {
            int m = tm + (half << 3) + r;
            float v = acc[j][r] + bv;
            if (RELU) v = fmaxf(v, 0.0f);
            C[(size_t)m * N + n] = v;
        }
    }
}

// ---------------- small utility kernels ----------------
__global__ void k_zero(float* p, long n) {
    long t = (long)blockIdx.x * blockDim.x + threadIdx.x;
    if (t < n) p[t] = 0.0f;
}

__global__ void k_deg(const int* __restrict__ dst, float* __restrict__ deg, int E) {
    int t = blockIdx.x * blockDim.x + threadIdx.x;
    if (t < E) atomicAdd(&deg[dst[t]], 1.0f);
}

__global__ void k_rsqrt(const float* __restrict__ deg, float* __restrict__ dinv, int n) {
    int t = blockIdx.x * blockDim.x + threadIdx.x;
    if (t < n) dinv[t] = rsqrtf(deg[t] + 1.0f);
}

// agg[dst] += dinv[src]*dinv[dst] * Ht[src]  — one thread per (edge, channel)
__global__ void k_agg(const int* __restrict__ src, const int* __restrict__ dst,
                      const float* __restrict__ dinv, const float* __restrict__ Ht,
                      float* __restrict__ agg, int E) {
    long t = (long)blockIdx.x * blockDim.x + threadIdx.x;
    if (t >= (long)E * 128) return;
    int e = (int)(t >> 7);
    int c = (int)(t & 127);
    int s = src[e], d = dst[e];
    float nrm = dinv[s] * dinv[d];
    atomicAdd(&agg[(size_t)d * 128 + c], nrm * Ht[(size_t)s * 128 + c]);
}

// out = relu(agg + dinv^2 * Ht + b)
__global__ void k_finish(const float* __restrict__ agg, const float* __restrict__ Ht,
                         const float* __restrict__ dinv, const float* __restrict__ b,
                         float* __restrict__ out, int total) {
    int t = blockIdx.x * blockDim.x + threadIdx.x;
    if (t >= total) return;
    int n = t >> 7, c = t & 127;
    float sn = dinv[n]; sn *= sn;
    out[t] = fmaxf(agg[t] + sn * Ht[t] + b[c], 0.0f);
}

__global__ void k_pool(const float* __restrict__ h, const int* __restrict__ batch,
                       float* __restrict__ sums, float* __restrict__ cnts, int total) {
    int t = blockIdx.x * blockDim.x + threadIdx.x;
    if (t >= total) return;
    int n = t >> 7, c = t & 127;
    int g = batch[n];
    atomicAdd(&sums[(size_t)g * 128 + c], h[t]);
    if (c == 0) atomicAdd(&cnts[g], 1.0f);
}

__global__ void k_pooldiv(const float* __restrict__ sums, const float* __restrict__ cnts,
                          float* __restrict__ pooled, int total) {
    int t = blockIdx.x * blockDim.x + threadIdx.x;
    if (t >= total) return;
    pooled[t] = sums[t] / fmaxf(cnts[t >> 7], 1.0f);
}

// ---------------- transposed 3D convs (k=4, s=2, p=1) ----------------
// output o receives taps k with o = 2*i + k - 1  ->  at most 2 valid (k,i) per dim
__device__ __forceinline__ int taps(int o, int ilim, int* kk, int* ii) {
    int n = 0;
    for (int k = (o + 1) & 1; k < 4; k += 2) {
        int i = (o + 1 - k) >> 1;
        if (i >= 0 && i < ilim) { kk[n] = k; ii[n] = i; ++n; }
    }
    return n;
}

// deconv1: [64,8,8,8] -> relu -> bf16 [32,16,16,16]; one thread per spatial pos, 32 oc accs
__global__ void k_deconv1(const float* __restrict__ hdec, const float* __restrict__ w,
                          const float* __restrict__ b, __bf16* __restrict__ t1) {
    int g = blockIdx.x >> 4;
    int s = ((blockIdx.x & 15) << 8) | threadIdx.x;      // 0..4095
    int od = s >> 8, oh = (s >> 4) & 15, ow = s & 15;
    int kd[2], id[2], kh[2], ih[2], kw[2], iw[2];
    int nd = taps(od, 8, kd, id), nh = taps(oh, 8, kh, ih), nw = taps(ow, 8, kw, iw);
    float acc[32];
#pragma unroll
    for (int oc = 0; oc < 32; ++oc) acc[oc] = b[oc];
    const float* in = hdec + (size_t)g * 32768;
    for (int ic = 0; ic < 64; ++ic) {
        const float* xc = in + ic * 512;
        const float* wc = w + (size_t)ic * 32 * 64;      // dw1[ic][oc][kd][kh][kw]
        for (int aa = 0; aa < nd; ++aa)
            for (int cc = 0; cc < nh; ++cc)
                for (int ee = 0; ee < nw; ++ee) {
                    float xv = xc[id[aa] * 64 + ih[cc] * 8 + iw[ee]];
                    int ko = kd[aa] * 16 + kh[cc] * 4 + kw[ee];
#pragma unroll
                    for (int oc = 0; oc < 32; ++oc)
                        acc[oc] = fmaf(xv, wc[(oc << 6) + ko], acc[oc]);
                }
    }
    __bf16* out = t1 + (size_t)g * 131072 + s;
#pragma unroll
    for (int oc = 0; oc < 32; ++oc)
        out[(size_t)oc * 4096] = (__bf16)fmaxf(acc[oc], 0.0f);
}

// deconv2: bf16 [32,16,16,16] -> softplus -> f32 [1,32,32,32]; one thread per output voxel
__global__ void k_deconv2(const __bf16* __restrict__ t1, const float* __restrict__ w,
                          const float* __restrict__ b, float* __restrict__ out) {
    int g = blockIdx.x >> 7;
    int s = ((blockIdx.x & 127) << 8) | threadIdx.x;     // 0..32767
    int od = s >> 10, oh = (s >> 5) & 31, ow = s & 31;
    int kd[2], id[2], kh[2], ih[2], kw[2], iw[2];
    int nd = taps(od, 16, kd, id), nh = taps(oh, 16, kh, ih), nw = taps(ow, 16, kw, iw);
    float acc = b[0];
    const __bf16* tg = t1 + (size_t)g * 131072;
    for (int ic = 0; ic < 32; ++ic) {
        const __bf16* xc = tg + ic * 4096;
        const float*  wc = w + ic * 64;                  // dw2[ic][0][kd][kh][kw]
        for (int aa = 0; aa < nd; ++aa)
            for (int cc = 0; cc < nh; ++cc)
                for (int ee = 0; ee < nw; ++ee)
                    acc = fmaf((float)xc[id[aa] * 256 + ih[cc] * 16 + iw[ee]],
                               wc[kd[aa] * 16 + kh[cc] * 4 + kw[ee]], acc);
    }
    float v = (acc > 20.0f) ? acc : log1pf(__expf(acc));
    out[(size_t)g * 32768 + s] = v;
}

// =====================================================================
extern "C" void kernel_launch(void* const* d_in, const int* in_sizes, int n_in,
                              void* d_out, int out_size, void* d_ws, size_t ws_size,
                              hipStream_t stream) {
    const float* x   = (const float*)d_in[0];
    const int*   ei  = (const int*)d_in[1];
    const int*   bat = (const int*)d_in[2];
    const float* W1 = (const float*)d_in[3];  const float* b1 = (const float*)d_in[4];
    const float* W2 = (const float*)d_in[5];  const float* b2 = (const float*)d_in[6];
    const float* W3 = (const float*)d_in[7];  const float* b3 = (const float*)d_in[8];
    const float* Wl = (const float*)d_in[9];  const float* bl = (const float*)d_in[10];
    const float* Wd = (const float*)d_in[11]; const float* bd = (const float*)d_in[12];
    const float* dw1 = (const float*)d_in[13]; const float* db1 = (const float*)d_in[14];
    const float* dw2 = (const float*)d_in[15]; const float* db2 = (const float*)d_in[16];

    int Nn = in_sizes[0] / 128;      // 50000
    int E  = in_sizes[1] / 2;        // 500000
    const int* src  = ei;
    const int* dstv = ei + E;

    // ---- workspace layout (floats); total ~29.77M floats (~119 MB) ----
    float*  ws     = (float*)d_ws;
    float*  deg    = ws;                    // 50000
    float*  dinv   = ws + 50000;            // 50000
    float*  T      = ws + 100000;           // 6.4M   (Ht per layer)
    float*  Ag     = ws + 6500000;          // 6.4M   (aggregation)
    float*  h1     = ws + 12900000;         // 6.4M
    float*  h2     = ws + 19300000;         // 6.4M
    float*  hdec   = T;                     // 8.39M floats, spans T+Ag (GCN done by then)
    __bf16* t1     = (__bf16*)(ws + 12900000); // 33.55M bf16, reuses h1/h2 (dead after pool)
    float*  sums   = ws + 29700000;         // 32768
    float*  cnts   = ws + 29732768;         // 256
    float*  pooled = ws + 29733024;         // 32768

    float* vox    = (float*)d_out;                    // [256,1,32,32,32]
    float* latent = vox + (size_t)256 * 32768;        // [256,256]

    auto nb = [](long n) { return (int)((n + 255) / 256); };
    auto gblk = [](int M, int N) {                    // blocks for 16x64 wave tiles
        long waves = (long)(M >> 4) * (N >> 6);
        return (int)((waves + 7) / 8);
    };

    long NC = (long)Nn * 128;

    // degrees & symmetric norm
    k_zero<<<nb(Nn), 256, 0, stream>>>(deg, Nn);
    k_deg<<<nb(E), 256, 0, stream>>>(dstv, deg, E);
    k_rsqrt<<<nb(Nn), 256, 0, stream>>>(deg, dinv, Nn);

    // 3 GCN layers
    const float* hin = x;
    const float* Wv[3] = {W1, W2, W3};
    const float* bv[3] = {b1, b2, b3};
    float* hout[3] = {h1, h2, h1};
    for (int l = 0; l < 3; ++l) {
        k_gemm<128, false, false><<<gblk(Nn, 128), 256, 0, stream>>>(
            hin, Wv[l], nullptr, T, Nn, 128);                        // WMMA transform
        k_zero<<<nb(NC), 256, 0, stream>>>(Ag, NC);
        k_agg<<<nb((long)E * 128), 256, 0, stream>>>(src, dstv, dinv, T, Ag, E);
        k_finish<<<nb(NC), 256, 0, stream>>>(Ag, T, dinv, bv[l], hout[l], (int)NC);
        hin = hout[l];
    }

    // global mean pool
    k_zero<<<nb(256 * 128), 256, 0, stream>>>(sums, 256 * 128);
    k_zero<<<1, 256, 0, stream>>>(cnts, 256);
    k_pool<<<nb(NC), 256, 0, stream>>>(hin, bat, sums, cnts, (int)NC);
    k_pooldiv<<<nb(256 * 128), 256, 0, stream>>>(sums, cnts, pooled, 256 * 128);

    // latent = pooled @ Wl + bl   (WMMA, M=256 N=256 K=128)
    k_gemm<128, true, false><<<gblk(256, 256), 256, 0, stream>>>(
        pooled, Wl, bl, latent, 256, 256);
    // hdec = latent @ Wd + bd     (WMMA, M=256 N=32768 K=256)
    k_gemm<256, true, false><<<gblk(256, 32768), 256, 0, stream>>>(
        latent, Wd, bd, hdec, 256, 32768);

    // decoder deconvs
    k_deconv1<<<256 * 16, 256, 0, stream>>>(hdec, dw1, db1, t1);
    k_deconv2<<<256 * 128, 256, 0, stream>>>(t1, dw2, db2, vox);
}
